// PatchNCELoss_bicut_5566277616526
// MI455X (gfx1250) — compile-verified, matrix operand order checked by reading, not run on hardware
//
#include <hip/hip_runtime.h>

// ---------------------------------------------------------------------------
// PatchNCE loss, fused:  loss[i] = logsumexp([l_pos/T, masked_sim_row/T]) - l_pos/T
//
// Pipeline:
//   1) row_norms_kernel      : ||q_i||, ||k_j||                    (ws)
//   2) normalize_split_kernel: x/||x|| -> bf16 hi + bf16 residual  (ws, 16MB)
//   3) nce_loss_kernel       : 8192x8192x256 GEMM of normalized vectors via
//      split-precision WMMA (hi*hi + hi*lo + lo*hi, f32 accum) == cosine sim,
//      with diag-mask + streaming logsumexp fused in the epilogue.
//      B tiles are staged in LDS (double-buffered, operand-order layout) and
//      shared by the 4 waves of the block.
// ---------------------------------------------------------------------------

#define NN   8192
#define DD   256

typedef __attribute__((ext_vector_type(16))) __bf16 v16bf;
typedef __attribute__((ext_vector_type(8)))  __bf16 v8bf;
typedef __attribute__((ext_vector_type(8)))  float  v8f;

constexpr float INV_T    = 1.0f / 0.07f;                              // 1/nce_T
constexpr float LN2      = 0.6931471805599453f;
constexpr float INV_T_L2 = (1.0f / 0.07f) * 1.4426950408889634f;      // (1/T)*log2(e)

union U16 { v16bf v; v8bf h[2]; };

// ---------------------------------------------------------------------------
// Kernel 1: row L2 norms for q (slots 0..N-1) and k (slots N..2N-1).
// ---------------------------------------------------------------------------
__global__ __launch_bounds__(256) void row_norms_kernel(
        const float* __restrict__ q, const float* __restrict__ k,
        float* __restrict__ norms) {
    const int gwave = (blockIdx.x * blockDim.x + threadIdx.x) >> 5;
    const int lane  = threadIdx.x & 31;
    const float* src = (gwave < NN) ? q : k;
    const int row    = (gwave < NN) ? gwave : gwave - NN;

    const float* p = src + (size_t)row * DD + lane * 8;
    float4 a = ((const float4*)p)[0];
    float4 b = ((const float4*)p)[1];
    float s = a.x * a.x + a.y * a.y + a.z * a.z + a.w * a.w +
              b.x * b.x + b.y * b.y + b.z * b.z + b.w * b.w;
#pragma unroll
    for (int off = 16; off >= 1; off >>= 1) s += __shfl_xor(s, off, 32);
    if (lane == 0) norms[gwave] = __builtin_amdgcn_sqrtf(s);
}

// ---------------------------------------------------------------------------
// Kernel 2: normalize rows and split into bf16 hi + bf16 residual lo.
// 8 contiguous elements per thread (16B bf16 stores).
// ---------------------------------------------------------------------------
__global__ __launch_bounds__(256) void normalize_split_kernel(
        const float* __restrict__ q, const float* __restrict__ k,
        const float* __restrict__ norms,
        __bf16* __restrict__ qhi, __bf16* __restrict__ qlo,
        __bf16* __restrict__ khi, __bf16* __restrict__ klo) {
    const size_t t  = (size_t)blockIdx.x * blockDim.x + threadIdx.x;
    const size_t e0 = t * 8;                        // [0, 2*NN*DD)
    const bool  isQ = e0 < (size_t)NN * DD;
    const size_t off = isQ ? e0 : e0 - (size_t)NN * DD;
    const int   row  = (int)(off / DD);

    const float rn = 1.0f / fmaxf(norms[(isQ ? 0 : NN) + row], 1e-30f);

    const float* src = (isQ ? q : k) + off;
    float4 a = ((const float4*)src)[0];
    float4 b = ((const float4*)src)[1];
    float v[8] = {a.x, a.y, a.z, a.w, b.x, b.y, b.z, b.w};

    v8bf hi, lo;
#pragma unroll
    for (int e = 0; e < 8; ++e) {
        float  x = v[e] * rn;
        __bf16 h = (__bf16)x;
        hi[e] = h;
        lo[e] = (__bf16)(x - (float)h);
    }
    *(v8bf*)((isQ ? qhi : khi) + off) = hi;
    *(v8bf*)((isQ ? qlo : klo) + off) = lo;
}

// ---------------------------------------------------------------------------
// Stage one K-chunk (32 elems = hi0,hi1,lo0,lo1 groups of 8) into LDS
// operand-order slot for this lane.
// ---------------------------------------------------------------------------
__device__ __forceinline__ void stage_chunk(const __bf16* __restrict__ hrow,
                                            const __bf16* __restrict__ lrow,
                                            int off, __bf16* dst) {
    v8bf h0 = *(const v8bf*)(hrow + off);
    v8bf h1 = *(const v8bf*)(hrow + off + 16);
    v8bf l0 = *(const v8bf*)(lrow + off);
    v8bf l1 = *(const v8bf*)(lrow + off + 16);
    ((v8bf*)dst)[0] = h0;
    ((v8bf*)dst)[1] = h1;
    ((v8bf*)dst)[2] = l0;
    ((v8bf*)dst)[3] = l1;
}

// ---------------------------------------------------------------------------
// Kernel 3: fused GEMM + diag-mask + streaming logsumexp.
// 128 threads = 4 waves; each wave owns one 16-row block; the block shares
// double-buffered B tiles in LDS (each wave stages 2 of 8 K-chunks).
// ---------------------------------------------------------------------------
__global__ __launch_bounds__(128) void nce_loss_kernel(
        const __bf16* __restrict__ qhi, const __bf16* __restrict__ qlo,
        const __bf16* __restrict__ khi, const __bf16* __restrict__ klo,
        float* __restrict__ out) {
    __shared__ __align__(16) __bf16 lds[2][8][32][32];   // 32 KB

    const int lane   = threadIdx.x & 31;
    const int wave   = threadIdx.x >> 5;
    const int rowblk = blockIdx.x * 4 + wave;            // 0..511
    const int base_r = rowblk * 16;
    const int col16  = lane & 15;                        // A row / B col / C col
    const int hi     = lane >> 4;                        // lane half
    const int kb     = hi * 8;                           // per-half K base

    // ---- resident A: 16 rows x 256 K of normalized q, bf16 hi/lo
    const __bf16* qh = qhi + (size_t)(base_r + col16) * DD;
    const __bf16* ql = qlo + (size_t)(base_r + col16) * DD;
    v16bf Ahi[8], Alo[8];
#pragma unroll
    for (int c = 0; c < 8; ++c) {
        U16 uh, ul;
        const int off = 32 * c + kb;
        uh.h[0] = *(const v8bf*)(qh + off);
        uh.h[1] = *(const v8bf*)(qh + off + 16);
        ul.h[0] = *(const v8bf*)(ql + off);
        ul.h[1] = *(const v8bf*)(ql + off + 16);
        Ahi[c] = uh.v;
        Alo[c] = ul.v;
    }

    float sum[8], posT[8];
#pragma unroll
    for (int r = 0; r < 8; ++r) { sum[r] = 0.0f; posT[r] = 0.0f; }

    constexpr int T = NN / 16;                           // 512 column tiles

    // ---- prologue: stage tile 0 into buffer 0
    {
        const __bf16* hr = khi + (size_t)col16 * DD;
        const __bf16* lr = klo + (size_t)col16 * DD;
#pragma unroll
        for (int i = 0; i < 2; ++i) {
            const int c = wave * 2 + i;
            stage_chunk(hr, lr, 32 * c + kb, &lds[0][c][lane][0]);
        }
    }
    __syncthreads();

    for (int ct = 0; ct < T; ++ct) {
        const int b = ct & 1;

        // ---- stage next tile into the other buffer
        if (ct + 1 < T) {
            const int nc = (ct + 1) * 16;
            const __bf16* hr = khi + (size_t)(nc + col16) * DD;
            const __bf16* lr = klo + (size_t)(nc + col16) * DD;
            __builtin_prefetch(hr + DD * 16, 0, 3);      // gfx1250 global_prefetch
#pragma unroll
            for (int i = 0; i < 2; ++i) {
                const int c = wave * 2 + i;
                stage_chunk(hr, lr, 32 * c + kb, &lds[b ^ 1][c][lane][0]);
            }
        }

        // ---- consume current tile from LDS
        v8f acc = {};
#pragma unroll
        for (int c = 0; c < 8; ++c) {
            U16 bh, bl;
            const __bf16* s = &lds[b][c][lane][0];
            bh.h[0] = ((const v8bf*)s)[0];
            bh.h[1] = ((const v8bf*)s)[1];
            bl.h[0] = ((const v8bf*)s)[2];
            bl.h[1] = ((const v8bf*)s)[3];
            acc = __builtin_amdgcn_wmma_f32_16x16x32_bf16(false, Ahi[c], false, bh.v,
                                                          (short)0, acc, false, false);
            acc = __builtin_amdgcn_wmma_f32_16x16x32_bf16(false, Ahi[c], false, bl.v,
                                                          (short)0, acc, false, false);
            acc = __builtin_amdgcn_wmma_f32_16x16x32_bf16(false, Alo[c], false, bh.v,
                                                          (short)0, acc, false, false);
        }

        // ---- epilogue: acc IS cosine similarity (vectors pre-normalized).
        // exp(sim/T) also serves as the col-0 (l_pos) term on the diagonal
        // tile; the masked diag term exp(-10/T) underflows f32 to 0 (as in
        // the reference's f32 logsumexp).
#pragma unroll
        for (int r = 0; r < 8; ++r)
            sum[r] += __builtin_amdgcn_exp2f(acc[r] * INV_T_L2);

        const int base_c = ct * 16;
        if (base_c == base_r) {                          // wave-uniform diag tile
            const int rd = col16 - kb;                   // r with row == col
#pragma unroll
            for (int r = 0; r < 8; ++r)
                if (rd == r) posT[r] = acc[r] * INV_T;
        }

        __syncthreads();
    }

    // ---- reduce across the 16 lanes of each half (columns); posT has exactly
    //      one nonzero contributor per row, so sum == value.
#pragma unroll
    for (int r = 0; r < 8; ++r) {
#pragma unroll
        for (int off = 1; off < 16; off <<= 1) {
            sum[r]  += __shfl_xor(sum[r], off, 32);
            posT[r] += __shfl_xor(posT[r], off, 32);
        }
    }
    if (col16 == 0) {
#pragma unroll
        for (int r = 0; r < 8; ++r)
            out[base_r + hi * 8 + r] =
                __builtin_amdgcn_logf(sum[r]) * LN2 - posT[r];
    }
}

// ---------------------------------------------------------------------------
extern "C" void kernel_launch(void* const* d_in, const int* in_sizes, int n_in,
                              void* d_out, int out_size, void* d_ws, size_t ws_size,
                              hipStream_t stream) {
    (void)in_sizes; (void)n_in; (void)out_size; (void)ws_size;
    const float* q = (const float*)d_in[0];
    const float* k = (const float*)d_in[1];
    float* out = (float*)d_out;

    // workspace layout: norms (64KB) | qhi | qlo | khi | klo  (4MB each)
    float*  norms = (float*)d_ws;
    __bf16* qhi   = (__bf16*)((char*)d_ws + 65536);
    __bf16* qlo   = qhi + (size_t)NN * DD;
    __bf16* khi   = qlo + (size_t)NN * DD;
    __bf16* klo   = khi + (size_t)NN * DD;

    row_norms_kernel<<<(2 * NN) / 8, 256, 0, stream>>>(q, k, norms);
    normalize_split_kernel<<<(2 * NN * DD / 8) / 256, 256, 0, stream>>>(
        q, k, norms, qhi, qlo, khi, klo);
    nce_loss_kernel<<<(NN / 16) / 4, 128, 0, stream>>>(qhi, qlo, khi, klo, out);
}
